// SSD_Net_84250078478895
// MI455X (gfx1250) — compile-verified, matrix operand order checked by reading, not run on hardware
//
#include <hip/hip_runtime.h>

typedef __attribute__((ext_vector_type(16))) __bf16 v16bf;
typedef __attribute__((ext_vector_type(8)))  float  v8f;
typedef __attribute__((ext_vector_type(4)))  int    v4i;

#define BATCH 128
#define NPRI  8732
#define CCH   512
#define HDIM  19
#define NH    4096
#define LROW  72   // LDS row stride in halves: 144B -> 16B aligned, conflict-free b128 reads

#if defined(__AMDGCN__) && __has_builtin(__builtin_amdgcn_global_load_async_to_lds_b128)
#define HAVE_ASYNC_LDS 1
#else
#define HAVE_ASYNC_LDS 0
#endif

__device__ __forceinline__ unsigned short f2bf(float f) {
    unsigned u = __float_as_uint(f);
    unsigned r = u + 0x7FFFu + ((u >> 16) & 1u);   // round-to-nearest-even
    return (unsigned short)(r >> 16);
}

union FragBF { v16bf v; uint4 q[2]; };

__device__ __forceinline__ void wait_async_lds() {
#if HAVE_ASYNC_LDS
#if __has_builtin(__builtin_amdgcn_s_wait_asynccnt)
    __builtin_amdgcn_s_wait_asynccnt(0);
#else
    asm volatile("s_wait_asynccnt 0x0" ::: "memory");
#endif
#endif
}

__device__ __forceinline__ void async_copy_b128(const unsigned short* g, unsigned short* l) {
#if HAVE_ASYNC_LDS
    __builtin_amdgcn_global_load_async_to_lds_b128(
        (__attribute__((address_space(1))) v4i*)(g),
        (__attribute__((address_space(3))) v4i*)(l), 0, 0);
#endif
}

// ---------------- Stage A: weight convert (+transpose) fp32 -> bf16 --------
// W [K x N] row-major fp32  ->  Wt [N x K] row-major bf16
__global__ void convT_kernel(const float* __restrict__ W, unsigned short* __restrict__ Wt,
                             int K, int N) {
    __shared__ unsigned short t[32][33];
    int n0 = blockIdx.x * 32, k0 = blockIdx.y * 32;
    int tx = threadIdx.x & 31, ty = threadIdx.x >> 5;     // 32 x 8
    for (int r = ty; r < 32; r += 8)
        t[r][tx] = f2bf(W[(size_t)(k0 + r) * N + n0 + tx]);
    __syncthreads();
    for (int r = ty; r < 32; r += 8)
        Wt[(size_t)(n0 + r) * K + k0 + tx] = t[tx][r];
}

// w3 [4096 x 19] -> wt3 [32 x 4096] bf16, rows 19..31 zero
__global__ void convW3_kernel(const float* __restrict__ w3, unsigned short* __restrict__ wt3) {
    int k = blockIdx.x * 256 + threadIdx.x;               // 0..4095
    for (int n = 0; n < 32; n++)
        wt3[(size_t)n * NH + k] = (n < HDIM) ? f2bf(w3[(size_t)k * HDIM + n]) : (unsigned short)0;
}

// ---------------- Stage B: best prior + decode -----------------------------
__global__ void best_prior_kernel(const float* __restrict__ loc, const float* __restrict__ conf,
                                  const float* __restrict__ priors, float* __restrict__ out,
                                  int* __restrict__ meta) {
    int b = blockIdx.x, t = threadIdx.x;
    __shared__ float ss[256];
    __shared__ int   si[256];
    float bs = -INFINITY; int bi = -1;
    for (int i = t; i < NPRI; i += 256) {
        float c0 = conf[((size_t)b * NPRI + i) * 2 + 0];
        float c1 = conf[((size_t)b * NPRI + i) * 2 + 1];
        if (c1 > c0) {                                    // argmax class == 1
            float s = c1 - c0;                            // monotone with softmax prob1
            if (s > bs) { bs = s; bi = i; }               // keep smallest index on tie
        }
    }
    ss[t] = bs; si[t] = bi;
    __syncthreads();
    for (int o = 128; o > 0; o >>= 1) {
        if (t < o) {
            bool take = (ss[t + o] > ss[t]) ||
                        (ss[t + o] == ss[t] && si[t + o] >= 0 && (si[t] < 0 || si[t + o] < si[t]));
            if (take) { ss[t] = ss[t + o]; si[t] = si[t + o]; }
        }
        __syncthreads();
    }
    if (t == 0) {
        int idx = si[0];
        int has = (idx >= 0);
        float bx[4] = {0.f, 0.f, 0.f, 0.f};
        int geom = 0;
        if (has) {
            size_t lb = ((size_t)b * NPRI + idx) * 4;
            float l0 = loc[lb + 0], l1 = loc[lb + 1], l2 = loc[lb + 2], l3 = loc[lb + 3];
            float px = priors[idx * 4 + 0], py = priors[idx * 4 + 1];
            float pw = priors[idx * 4 + 2], ph = priors[idx * 4 + 3];
            float cx = px + l0 * 0.1f * pw;
            float cy = py + l1 * 0.1f * ph;
            float w  = pw * expf(l2 * 0.2f);
            float h  = ph * expf(l3 * 0.2f);
            float x1 = cx - 0.5f * w, y1 = cy - 0.5f * h;
            float x2 = x1 + w,        y2 = y1 + h;
            geom = (x2 > x1) && (y2 > y1);
            bx[0] = x1; bx[1] = y1; bx[2] = x2; bx[3] = y2;
        }
        int valid = has && geom;
        float res[4];
        for (int j = 0; j < 4; j++) {
            float v = has ? bx[j] : 0.f;
            v = fminf(fmaxf(v, 0.f), 1.f);
            res[j] = v;
            out[2432 + b * 4 + j] = v;                    // boxes output
        }
        ((int*)out)[2944 + b] = valid;                    // valid output (int32 bits)
        const float step = 300.0f / (float)HDIM;
        for (int j = 0; j < 4; j++) {
            float v = fminf(fmaxf(res[j] * 300.f, 0.f), 300.f) / step;
            v = fminf(fmaxf(v, 0.f), (float)(HDIM - 1));
            meta[b * 5 + j] = (int)floorf(v);
        }
        meta[b * 5 + 4] = valid;
    }
}

// ---------------- Stage C: crop + max-pool -> bf16 feat --------------------
__global__ void extract_kernel(const float* __restrict__ features, const int* __restrict__ meta,
                               unsigned short* __restrict__ feat) {
    int b = blockIdx.x;
    int x1 = meta[b * 5 + 0], y1 = meta[b * 5 + 1];
    int x2 = meta[b * 5 + 2], y2 = meta[b * 5 + 3];
    int valid = meta[b * 5 + 4];
    for (int c = threadIdx.x; c < CCH; c += blockDim.x) {
        const float* f = features + ((size_t)b * CCH + c) * (HDIM * HDIM);
        float m = -INFINITY;
        for (int y = y1; y <= y2; y++)
            for (int x = x1; x <= x2; x++)
                m = fmaxf(m, f[y * HDIM + x]);
        feat[(size_t)b * CCH + c] = f2bf(valid ? m : 0.f);
    }
}

// ---------------- Stage D: WMMA GEMM helpers -------------------------------
template<int NTILE>
__device__ __forceinline__ void issue_async_chunk(const unsigned short* __restrict__ Wt, int K,
                                                  int n0, int k0, unsigned short* buf, int tid) {
    constexpr int TOT = NTILE * 4;                        // uint4 tiles per chunk
#pragma unroll
    for (int it = 0; it < (TOT + 255) / 256; it++) {
        int i = tid + it * 256;
        if ((TOT % 256) == 0 || i < TOT) {
            int r = i >> 2, p = i & 3;
            async_copy_b128(Wt + (size_t)(n0 + r) * K + k0 + p * 8, buf + r * LROW + p * 8);
        }
    }
}

template<int NTILE, int ITERS>
__device__ __forceinline__ void load_chunk(const unsigned short* __restrict__ Wt, int K,
                                           int n0, int k0, int tid, uint4* regs) {
#pragma unroll
    for (int it = 0; it < ITERS; it++) {
        int i = tid + it * 256;
        if ((NTILE * 4) % 256 == 0 || i < NTILE * 4) {
            int r = i >> 2, p = i & 3;
            regs[it] = *(const uint4*)(Wt + (size_t)(n0 + r) * K + k0 + p * 8);
        }
    }
}

template<int NTILE, int ITERS>
__device__ __forceinline__ void store_chunk(unsigned short* buf, int tid, const uint4* regs) {
#pragma unroll
    for (int it = 0; it < ITERS; it++) {
        int i = tid + it * 256;
        if ((NTILE * 4) % 256 == 0 || i < NTILE * 4) {
            int r = i >> 2, p = i & 3;
            *(uint4*)&buf[r * LROW + p * 8] = regs[it];
        }
    }
}

// ---------------- Stage D: WMMA GEMM  out = relu(A @ W + b) ----------------
// A [128 x K] bf16 row-major, Wt [N x K] bf16 row-major (pre-transposed W)
// 8 waves: wave w -> rows 16w..16w+15. NSUB 16-col groups per workgroup.
// Double-buffered LDS pipeline: one barrier per k-step; next Wt chunk is
// fetched (async-to-LDS when available) while current feeds the WMMAs.
template<int NSUB, bool RELU, bool OUTF32>
__global__ __launch_bounds__(256)
void gemm_kernel(const unsigned short* __restrict__ A, const unsigned short* __restrict__ Wt,
                 const float* __restrict__ bias, void* __restrict__ outp,
                 int K, int Nlog, int ldout) {
    constexpr int NTILE = NSUB * 16;
    constexpr int ITERS = (NTILE * 4 + 255) / 256;
    __shared__ unsigned short lds[2 * NTILE * LROW];

    const int tid  = threadIdx.x;
    const int lane = tid & 31;
    const int wv   = tid >> 5;
    const int n0   = blockIdx.x * NTILE;

    v8f acc[NSUB];
#pragma unroll
    for (int i = 0; i < NSUB; i++)
#pragma unroll
        for (int j = 0; j < 8; j++) acc[i][j] = 0.f;

    const int row   = wv * 16 + (lane & 15);              // A-matrix row for this lane
    const int khalf = (lane >> 4) * 8;                    // 16-bit fragment k-offset
    const int steps = K >> 5;

#if HAVE_ASYNC_LDS
    issue_async_chunk<NTILE>(Wt, K, n0, 0, lds, tid);
#else
    uint4 regs[ITERS];
    load_chunk<NTILE, ITERS>(Wt, K, n0, 0, tid, regs);
    store_chunk<NTILE, ITERS>(lds, tid, regs);
#endif

    int cur = 0;
    for (int s = 0; s < steps; s++) {
        const int k0 = s << 5;
        wait_async_lds();                                 // our copies into buf[cur] done
        __syncthreads();                                  // everyone's copies visible
        unsigned short* bufc = lds + cur * (NTILE * LROW);
        unsigned short* bufn = lds + (cur ^ 1) * (NTILE * LROW);

#if HAVE_ASYNC_LDS
        if (s + 1 < steps)
            issue_async_chunk<NTILE>(Wt, K, n0, k0 + 32, bufn, tid);
#else
        if (s + 1 < steps)
            load_chunk<NTILE, ITERS>(Wt, K, n0, k0 + 32, tid, regs);
#endif

        // A fragment: two contiguous b128 loads per lane
        FragBF a;
        const unsigned short* ap = A + (size_t)row * K + k0 + khalf;
        a.q[0] = *(const uint4*)ap;
        a.q[1] = *(const uint4*)(ap + 16);

        // load ALL B fragments first, then issue the WMMAs (overlap ds with matrix)
        FragBF bf[NSUB];
#pragma unroll
        for (int ns = 0; ns < NSUB; ns++) {
            const unsigned short* bp = bufc + (ns * 16 + (lane & 15)) * LROW + khalf;
            bf[ns].q[0] = *(const uint4*)bp;
            bf[ns].q[1] = *(const uint4*)(bp + 16);
        }
#pragma unroll
        for (int ns = 0; ns < NSUB; ns++)
            acc[ns] = __builtin_amdgcn_wmma_f32_16x16x32_bf16(
                false, a.v, false, bf[ns].v, (short)0, acc[ns], false, false);

#if !HAVE_ASYNC_LDS
        if (s + 1 < steps)
            store_chunk<NTILE, ITERS>(bufn, tid, regs);
#endif
        cur ^= 1;
    }

    // store: lane holds D rows rbase..rbase+7, col = n0 + ns*16 + (lane&15)
    const int rbase = wv * 16 + ((lane >> 4) * 8);
    const int cin   = lane & 15;
#pragma unroll
    for (int ns = 0; ns < NSUB; ns++) {
        int n = n0 + ns * 16 + cin;
        float bv = (n < Nlog) ? bias[n] : 0.f;
#pragma unroll
        for (int r = 0; r < 8; r++) {
            float v = acc[ns][r] + bv;
            if (RELU) v = fmaxf(v, 0.f);
            int rr = rbase + r;
            if (OUTF32) {
                if (n < Nlog) ((float*)outp)[(size_t)rr * ldout + n] = v;
            } else {
                ((unsigned short*)outp)[(size_t)rr * ldout + n] = f2bf(v);
            }
        }
    }
}

// ---------------------------------------------------------------------------
extern "C" void kernel_launch(void* const* d_in, const int* in_sizes, int n_in,
                              void* d_out, int out_size, void* d_ws, size_t ws_size,
                              hipStream_t stream) {
    const float* loc      = (const float*)d_in[0];
    const float* conf     = (const float*)d_in[1];
    const float* priors   = (const float*)d_in[2];
    const float* features = (const float*)d_in[3];
    const float* w1 = (const float*)d_in[4];
    const float* b1 = (const float*)d_in[5];
    const float* w2 = (const float*)d_in[6];
    const float* b2 = (const float*)d_in[7];
    const float* w3 = (const float*)d_in[8];
    const float* b3 = (const float*)d_in[9];
    float* out = (float*)d_out;

    char* ws = (char*)d_ws;
    size_t o_wt1  = 0;                                     // 4096*512*2  = 4 MiB
    size_t o_wt2  = o_wt1 + (size_t)NH * CCH * 2;          // 4096*4096*2 = 32 MiB
    size_t o_wt3  = o_wt2 + (size_t)NH * NH * 2;           // 32*4096*2
    size_t o_feat = o_wt3 + (size_t)32 * NH * 2;           // 128*512*2
    size_t o_h1   = o_feat + (size_t)BATCH * CCH * 2;      // 128*4096*2
    size_t o_h2   = o_h1 + (size_t)BATCH * NH * 2;         // 128*4096*2
    size_t o_meta = o_h2 + (size_t)BATCH * NH * 2;         // 128*5*4

    unsigned short* wt1  = (unsigned short*)(ws + o_wt1);
    unsigned short* wt2  = (unsigned short*)(ws + o_wt2);
    unsigned short* wt3  = (unsigned short*)(ws + o_wt3);
    unsigned short* feat = (unsigned short*)(ws + o_feat);
    unsigned short* h1   = (unsigned short*)(ws + o_h1);
    unsigned short* h2   = (unsigned short*)(ws + o_h2);
    int* meta            = (int*)(ws + o_meta);

    // weight conversion (+transpose)
    convT_kernel<<<dim3(NH / 32, CCH / 32), 256, 0, stream>>>(w1, wt1, CCH, NH);
    convT_kernel<<<dim3(NH / 32, NH / 32), 256, 0, stream>>>(w2, wt2, NH, NH);
    convW3_kernel<<<NH / 256, 256, 0, stream>>>(w3, wt3);

    // detection head
    best_prior_kernel<<<BATCH, 256, 0, stream>>>(loc, conf, priors, out, meta);
    extract_kernel<<<BATCH, 256, 0, stream>>>(features, meta, feat);

    // MLP via bf16 WMMA
    gemm_kernel<8, true,  false><<<NH / 128, 256, 0, stream>>>(feat, wt1, b1, h1, CCH, NH, NH);
    gemm_kernel<8, true,  false><<<NH / 128, 256, 0, stream>>>(h1,   wt2, b2, h2, NH,  NH, NH);
    gemm_kernel<2, false, true ><<<1,        256, 0, stream>>>(h2,   wt3, b3, out, NH, HDIM, HDIM);
}
